// ScaledDotProductAttention_15564961481220
// MI455X (gfx1250) — compile-verified, matrix operand order checked by reading, not run on hardware
//
#include <hip/hip_runtime.h>

// ---------------------------------------------------------------------------
// MI455X (gfx1250) multi-head attention, bf16 WMMA + fp32 accumulation.
// B=4, NQ=NK=2048, D=1024, H=16, head_dim=64.
// ---------------------------------------------------------------------------

typedef __attribute__((ext_vector_type(16))) __bf16 v16bf;
typedef __attribute__((ext_vector_type(8)))  float  v8f;

#define DMODEL 1024
#define NSEQ   2048
#define NHEAD  16
#define HDIM   64
#define MTOT   (4 * 2048)   // B * NQ rows

__device__ __forceinline__ v8f vzero() {
  v8f z = {0.f, 0.f, 0.f, 0.f, 0.f, 0.f, 0.f, 0.f};
  return z;
}

__device__ __forceinline__ v8f wmma_bf16(v16bf a, v16bf b, v8f c) {
  // D = A(16x32) * B(32x16) + C, fp32 accumulate
  return __builtin_amdgcn_wmma_f32_16x16x32_bf16(
      /*neg_a=*/false, a, /*neg_b=*/false, b,
      /*c_mod=*/(short)0, c, /*reuse_a=*/false, /*reuse_b=*/false);
}

// K index pattern for 16-bit A/B fragments (ISA 7.12.2, 16-bit A 16x32):
// lanes 0-15: vgpr v<4 -> K=2v,2v+1 ; v>=4 -> K=16+2(v-4)   (+8 for lanes 16-31)
__device__ __forceinline__ int kpat(int v, int hi) {
  return (v < 4 ? 2 * v : 16 + 2 * (v - 4)) + hi * 8;
}

// A-fragment from row-major storage: element (m,k) = base[m*ld + k].
// Also serves as B-fragment when B is stored N-major (i.e. B^T row-major).
__device__ __forceinline__ v16bf frag_rm_bf16(const __bf16* base, int ld) {
  const int lane = threadIdx.x & 31;
  const int m = lane & 15, hi = lane >> 4;
  const __bf16* row = base + (size_t)m * ld;
  v16bf f;
#pragma unroll
  for (int v = 0; v < 8; ++v) {
    const int k = kpat(v, hi);
    f[2 * v]     = row[k];
    f[2 * v + 1] = row[k + 1];
  }
  return f;
}

// Same, but fp32 source with on-the-fly bf16 conversion.
__device__ __forceinline__ v16bf frag_rm_f32(const float* base, int ld) {
  const int lane = threadIdx.x & 31;
  const int m = lane & 15, hi = lane >> 4;
  const float* row = base + (size_t)m * ld;
  v16bf f;
#pragma unroll
  for (int v = 0; v < 8; ++v) {
    const int k = kpat(v, hi);
    f[2 * v]     = (__bf16)row[k];
    f[2 * v + 1] = (__bf16)row[k + 1];
  }
  return f;
}

// Column-major access: element (m,k) = base[k*ld + m].
// A from transposed storage, or B from row-major KxN storage (pass base+n0).
__device__ __forceinline__ v16bf frag_cm_bf16(const __bf16* base, int ld) {
  const int lane = threadIdx.x & 31;
  const int m = lane & 15, hi = lane >> 4;
  v16bf f;
#pragma unroll
  for (int v = 0; v < 8; ++v) {
    const int k = kpat(v, hi);
    f[2 * v]     = base[(size_t)k * ld + m];
    f[2 * v + 1] = base[(size_t)(k + 1) * ld + m];
  }
  return f;
}

// CDNA5 async global->LDS copy (tracked by ASYNCcnt).
__device__ __forceinline__ void async_b128(unsigned lds_off, const void* gaddr) {
  asm volatile("global_load_async_to_lds_b128 %0, %1, off"
               :: "v"(lds_off), "v"((unsigned long long)gaddr)
               : "memory");
}

// ---------------------------------------------------------------------------
// Kernel 0: weight transpose + fp32->bf16.  WT[n*D + k] = W[k*D + n]
// ---------------------------------------------------------------------------
__global__ void wtrans_kernel(const float* __restrict__ W, __bf16* __restrict__ WT) {
  const size_t i = (size_t)blockIdx.x * 256 + threadIdx.x;  // over D*D
  const int k = (int)(i >> 10);
  const int n = (int)(i & 1023);
  WT[(size_t)n * DMODEL + k] = (__bf16)W[i];
}

// ---------------------------------------------------------------------------
// Kernel 1: projections Q/K/V = X * W  (bf16 out).  64x64 tile per wave.
// blockIdx.y selects which projection.  2048 tiles -> grid.x=512, 4 waves/WG.
// ---------------------------------------------------------------------------
__global__ __launch_bounds__(128) void proj_kernel(
    const float* __restrict__ Xq, const float* __restrict__ Xk, const float* __restrict__ Xv,
    const __bf16* __restrict__ WqT, const __bf16* __restrict__ WkT, const __bf16* __restrict__ WvT,
    __bf16* __restrict__ Qb, __bf16* __restrict__ Kb, __bf16* __restrict__ Vb) {
  const float* X; const __bf16* WT; __bf16* C;
  if (blockIdx.y == 0)      { X = Xq; WT = WqT; C = Qb; }
  else if (blockIdx.y == 1) { X = Xk; WT = WkT; C = Kb; }
  else                      { X = Xv; WT = WvT; C = Vb; }

  const int wid  = threadIdx.x >> 5;
  const int lane = threadIdx.x & 31;
  const int tile = blockIdx.x * 4 + wid;          // 0..2047
  const int m0 = (tile >> 4) * 64;                // 128 M-tiles
  const int n0 = (tile & 15) * 64;                // 16 N-tiles

  v8f acc[4][4];
#pragma unroll
  for (int i = 0; i < 4; ++i)
#pragma unroll
    for (int j = 0; j < 4; ++j) acc[i][j] = vzero();

  for (int k0 = 0; k0 < DMODEL; k0 += 32) {
    v16bf a[4], b[4];
#pragma unroll
    for (int i = 0; i < 4; ++i)
      a[i] = frag_rm_f32(X + (size_t)(m0 + 16 * i) * DMODEL + k0, DMODEL);
#pragma unroll
    for (int j = 0; j < 4; ++j)
      b[j] = frag_rm_bf16(WT + (size_t)(n0 + 16 * j) * DMODEL + k0, DMODEL);
#pragma unroll
    for (int i = 0; i < 4; ++i)
#pragma unroll
      for (int j = 0; j < 4; ++j)
        acc[i][j] = wmma_bf16(a[i], b[j], acc[i][j]);
  }

  const int n = lane & 15, hi = lane >> 4;
#pragma unroll
  for (int i = 0; i < 4; ++i)
#pragma unroll
    for (int j = 0; j < 4; ++j)
#pragma unroll
      for (int r = 0; r < 8; ++r) {
        const size_t row = (size_t)(m0 + 16 * i + r + hi * 8);
        const int    col = n0 + 16 * j + n;
        C[row * DMODEL + col] = (__bf16)acc[i][j][r];
      }
}

// ---------------------------------------------------------------------------
// Kernel 2: flash attention.  One WG (4 waves) per (b, h, 64-query tile).
// S^T = K*Q^T so softmax dim (keys) is the register dim; K/V tiles streamed
// into LDS with async b128 copies, double-buffered.
// ---------------------------------------------------------------------------
__global__ __launch_bounds__(128) void attn_kernel(
    const __bf16* __restrict__ Qb, const __bf16* __restrict__ Kb,
    const __bf16* __restrict__ Vb, __bf16* __restrict__ Ob) {
  __shared__ __bf16 sK[2][32 * 64];   // 32 keys x 64 head-dim, double buffered
  __shared__ __bf16 sV[2][32 * 64];
  __shared__ __bf16 sP[4][32 * 16];   // per-wave P^T tile (keys x queries)

  const int bid = blockIdx.x;
  const int qt  = bid & 31;
  const int h   = (bid >> 5) & 15;
  const int b   = bid >> 9;
  const int tid  = threadIdx.x;
  const int wid  = tid >> 5;
  const int lane = tid & 31;
  const int n  = lane & 15;
  const int hi = lane >> 4;

  const size_t rowQ = (size_t)(b * NSEQ + qt * 64 + wid * 16);
  const __bf16* Qbase = Qb + rowQ * DMODEL + h * HDIM;
  // B-operand = Q^T : Q is N-major for its columns -> row-major loader
  const v16bf bq0 = frag_rm_bf16(Qbase, DMODEL);
  const v16bf bq1 = frag_rm_bf16(Qbase + 32, DMODEL);

  const __bf16* Kh = Kb + (size_t)b * NSEQ * DMODEL + h * HDIM;
  const __bf16* Vh = Vb + (size_t)b * NSEQ * DMODEL + h * HDIM;

  v8f o0 = vzero(), o1 = vzero(), o2 = vzero(), o3 = vzero();
  float mrun = -__builtin_inff();
  float lrun = 0.f;

  const unsigned kbase0 = (unsigned)(unsigned long long)&sK[0][0];
  const unsigned vbase0 = (unsigned)(unsigned long long)&sV[0][0];

  // issue one 32x64 K tile + V tile: 512 x 16B segments / 128 threads = 4 asm
  // instructions per thread -> ASYNCcnt += 4 per wave per tile.
  auto issue_tile = [&](int t, int buf) {
    const __bf16* ksrc = Kh + (size_t)t * 32 * DMODEL;
    const __bf16* vsrc = Vh + (size_t)t * 32 * DMODEL;
    const unsigned kb = kbase0 + (unsigned)(buf * 32 * 64 * 2);
    const unsigned vb = vbase0 + (unsigned)(buf * 32 * 64 * 2);
#pragma unroll
    for (int s = 0; s < 2; ++s) {
      const int seg = tid + 128 * s;          // 0..255
      const int row = seg >> 3, c8 = seg & 7; // 8 bf16 per 16B segment
      async_b128(kb + (unsigned)seg * 16, ksrc + (size_t)row * DMODEL + c8 * 8);
      async_b128(vb + (unsigned)seg * 16, vsrc + (size_t)row * DMODEL + c8 * 8);
    }
  };

  const int NT = NSEQ / 32;  // 64 key tiles
  issue_tile(0, 0);

  for (int t = 0; t < NT; ++t) {
    const int buf = t & 1;
    if (t + 1 < NT) {
      issue_tile(t + 1, buf ^ 1);
      asm volatile("s_wait_asynccnt 0x4" ::: "memory");  // oldest tile done
    } else {
      asm volatile("s_wait_asynccnt 0x0" ::: "memory");
    }
    __syncthreads();

    const __bf16* tK = &sK[buf][0];
    const __bf16* tV = &sV[buf][0];
    __bf16* myP = &sP[wid][0];

    // S^T = K * Q^T  (M=32 keys as two 16-frags, N=16 queries, K-dim=64)
    v16bf a00 = frag_rm_bf16(tK, 64);
    v16bf a01 = frag_rm_bf16(tK + 32, 64);
    v16bf a10 = frag_rm_bf16(tK + 16 * 64, 64);
    v16bf a11 = frag_rm_bf16(tK + 16 * 64 + 32, 64);
    v8f s0 = vzero(), s1 = vzero();
    s0 = wmma_bf16(a00, bq0, s0);
    s0 = wmma_bf16(a01, bq1, s0);
    s1 = wmma_bf16(a10, bq0, s1);
    s1 = wmma_bf16(a11, bq1, s1);

    // online softmax over keys; reference scales by 1/sqrt(D_K)=1/32
    float tmax = -__builtin_inff();
#pragma unroll
    for (int r = 0; r < 8; ++r) {
      s0[r] *= 0.03125f;
      s1[r] *= 0.03125f;
      tmax = fmaxf(tmax, fmaxf(s0[r], s1[r]));
    }
    tmax = fmaxf(tmax, __shfl_xor(tmax, 16, 32));
    const float mnew = fmaxf(mrun, tmax);
    const float corr = __expf(mrun - mnew);
    float lsum = 0.f;
    float p0[8], p1[8];
#pragma unroll
    for (int r = 0; r < 8; ++r) {
      p0[r] = __expf(s0[r] - mnew);
      p1[r] = __expf(s1[r] - mnew);
      lsum += p0[r] + p1[r];
    }
    lsum += __shfl_xor(lsum, 16, 32);
    lrun = lrun * corr + lsum;
    mrun = mnew;

    // spill P^T (keys x 16 queries) to this wave's LDS slice, bf16
#pragma unroll
    for (int r = 0; r < 8; ++r) {
      myP[(r + hi * 8) * 16 + n]      = (__bf16)p0[r];
      myP[(16 + r + hi * 8) * 16 + n] = (__bf16)p1[r];
    }

    // rescale O accumulators; query index in O-layout is r + hi*8
    float fac[8];
#pragma unroll
    for (int r = 0; r < 8; ++r) fac[r] = __shfl(corr, r + hi * 8, 32);
#pragma unroll
    for (int r = 0; r < 8; ++r) {
      o0[r] *= fac[r]; o1[r] *= fac[r]; o2[r] *= fac[r]; o3[r] *= fac[r];
    }

    // O += P * V : A = P (16 queries x 32 keys) from LDS P^T (col-major),
    // B = V tile (32 keys x 64 head-dim, row-major in LDS)
    v16bf ap  = frag_cm_bf16(myP, 16);
    v16bf bv0 = frag_cm_bf16(tV + 0, 64);
    v16bf bv1 = frag_cm_bf16(tV + 16, 64);
    v16bf bv2 = frag_cm_bf16(tV + 32, 64);
    v16bf bv3 = frag_cm_bf16(tV + 48, 64);
    o0 = wmma_bf16(ap, bv0, o0);
    o1 = wmma_bf16(ap, bv1, o1);
    o2 = wmma_bf16(ap, bv2, o2);
    o3 = wmma_bf16(ap, bv3, o3);

    __syncthreads();  // all waves done with buf before it is overwritten
  }

  const float linv = 1.0f / lrun;
  float fn[8];
#pragma unroll
  for (int r = 0; r < 8; ++r) fn[r] = __shfl(linv, r + hi * 8, 32);
  __bf16* Obase = Ob + rowQ * DMODEL + h * HDIM;
#pragma unroll
  for (int r = 0; r < 8; ++r) {
    const size_t ro = (size_t)(r + hi * 8) * DMODEL;
    Obase[ro + 0 * 16 + n] = (__bf16)(o0[r] * fn[r]);
    Obase[ro + 1 * 16 + n] = (__bf16)(o1[r] * fn[r]);
    Obase[ro + 2 * 16 + n] = (__bf16)(o2[r] * fn[r]);
    Obase[ro + 3 * 16 + n] = (__bf16)(o3[r] * fn[r]);
  }
}

// ---------------------------------------------------------------------------
// Kernel 3: Y = O * Wo + bo  (fp32 out)
// ---------------------------------------------------------------------------
__global__ __launch_bounds__(128) void outproj_kernel(
    const __bf16* __restrict__ Ob, const __bf16* __restrict__ WoT,
    const float* __restrict__ bo, float* __restrict__ Y) {
  const int wid  = threadIdx.x >> 5;
  const int lane = threadIdx.x & 31;
  const int tile = blockIdx.x * 4 + wid;
  const int m0 = (tile >> 4) * 64;
  const int n0 = (tile & 15) * 64;

  v8f acc[4][4];
#pragma unroll
  for (int i = 0; i < 4; ++i)
#pragma unroll
    for (int j = 0; j < 4; ++j) acc[i][j] = vzero();

  for (int k0 = 0; k0 < DMODEL; k0 += 32) {
    v16bf a[4], b[4];
#pragma unroll
    for (int i = 0; i < 4; ++i)
      a[i] = frag_rm_bf16(Ob + (size_t)(m0 + 16 * i) * DMODEL + k0, DMODEL);
#pragma unroll
    for (int j = 0; j < 4; ++j)
      b[j] = frag_rm_bf16(WoT + (size_t)(n0 + 16 * j) * DMODEL + k0, DMODEL);
#pragma unroll
    for (int i = 0; i < 4; ++i)
#pragma unroll
      for (int j = 0; j < 4; ++j)
        acc[i][j] = wmma_bf16(a[i], b[j], acc[i][j]);
  }

  const int n = lane & 15, hi = lane >> 4;
#pragma unroll
  for (int i = 0; i < 4; ++i)
#pragma unroll
    for (int j = 0; j < 4; ++j)
#pragma unroll
      for (int r = 0; r < 8; ++r) {
        const size_t row = (size_t)(m0 + 16 * i + r + hi * 8);
        const int    col = n0 + 16 * j + n;
        Y[row * DMODEL + col] = acc[i][j][r] + bo[col];
      }
}

// ---------------------------------------------------------------------------
extern "C" void kernel_launch(void* const* d_in, const int* in_sizes, int n_in,
                              void* d_out, int out_size, void* d_ws, size_t ws_size,
                              hipStream_t stream) {
  (void)in_sizes; (void)n_in; (void)out_size; (void)ws_size;

  const float* q  = (const float*)d_in[0];
  const float* k  = (const float*)d_in[1];
  const float* v  = (const float*)d_in[2];
  const float* Wq = (const float*)d_in[3];
  const float* Wk = (const float*)d_in[4];
  const float* Wv = (const float*)d_in[5];
  const float* Wo = (const float*)d_in[6];
  const float* bo = (const float*)d_in[7];
  float* out = (float*)d_out;

  const size_t NEL = (size_t)MTOT * DMODEL;      // 8M elements
  const size_t WEL = (size_t)DMODEL * DMODEL;    // 1M elements
  __bf16* Qb  = (__bf16*)d_ws;
  __bf16* Kb  = Qb + NEL;
  __bf16* Vb  = Kb + NEL;
  __bf16* Obm = Vb + NEL;
  __bf16* WqT = Obm + NEL;
  __bf16* WkT = WqT + WEL;
  __bf16* WvT = WkT + WEL;
  __bf16* WoT = WvT + WEL;   // total: 4*16MB + 4*2MB = 72MB of workspace

  wtrans_kernel<<<4096, 256, 0, stream>>>(Wq, WqT);
  wtrans_kernel<<<4096, 256, 0, stream>>>(Wk, WkT);
  wtrans_kernel<<<4096, 256, 0, stream>>>(Wv, WvT);
  wtrans_kernel<<<4096, 256, 0, stream>>>(Wo, WoT);

  proj_kernel<<<dim3(512, 3, 1), 128, 0, stream>>>(q, k, v, WqT, WkT, WvT, Qb, Kb, Vb);

  attn_kernel<<<2048, 128, 0, stream>>>(Qb, Kb, Vb, Obm);

  outproj_kernel<<<512, 128, 0, stream>>>(Obm, WoT, bo, out);
}
